// DynamicGraphUnet_54657753809357
// MI455X (gfx1250) — matrix operand order repaired and checked by reading, not compiled, not run on hardware
//
#include <hip/hip_runtime.h>

// ---------------------------------------------------------------------------
// CDNA5 (gfx1250) types
// ---------------------------------------------------------------------------
typedef __bf16 bf16;
typedef __attribute__((ext_vector_type(16))) __bf16 bf16x16;
typedef __attribute__((ext_vector_type(8)))  __bf16 bf16x8;
typedef __attribute__((ext_vector_type(8)))  float  v8f;

static constexpr int N0 = 40962, N1 = 10242;   // fine / coarse vertex counts
static constexpr int E0 = 245760, E1 = 61440;  // fine / coarse edge counts
static constexpr float SLOPE = 0.3f;

__host__ __device__ static inline int ceil_div(int a, int b) { return (a + b - 1) / b; }

// ---------------------------------------------------------------------------
// degw[a] = sum of w_e over edges with source a
// ---------------------------------------------------------------------------
__global__ void k_degw(const int* __restrict__ ea, const float* __restrict__ w,
                       float* __restrict__ degw, int E) {
    int e = blockIdx.x * blockDim.x + threadIdx.x;
    if (e < E) atomicAdd(&degw[ea[e]], w[e]);
}

// ---------------------------------------------------------------------------
// agg[ea[e]][c] += w[e] * h[eb[e]][c]  (4 channels / thread, float4 gather)
// ---------------------------------------------------------------------------
__global__ void k_edge_agg(const int* __restrict__ ea, const int* __restrict__ eb,
                           const float* __restrict__ w, const float* __restrict__ h,
                           float* __restrict__ agg, int E, int C) {
    int c4 = C >> 2;
    int t = blockIdx.x * blockDim.x + threadIdx.x;
    if (t >= E * c4) return;
    int e = t / c4;
    int c = (t - e * c4) * 4;
    float we = w[e];
    const float4 hv = *reinterpret_cast<const float4*>(h + (size_t)eb[e] * C + c);
    float* d = agg + (size_t)ea[e] * C + c;
    atomicAdd(d + 0, we * hv.x);
    atomicAdd(d + 1, we * hv.y);
    atomicAdd(d + 2, we * hv.z);
    atomicAdd(d + 3, we * hv.w);
}

// ---------------------------------------------------------------------------
// Xin[v][k] = bf16( k<C ? degw[v]*h[v][k] : agg[v][k-C] )   (N x 2C, row major)
// ---------------------------------------------------------------------------
__global__ void k_build_xin(const float* __restrict__ h, const float* __restrict__ agg,
                            const float* __restrict__ degw, bf16* __restrict__ xin,
                            int Nv, int C) {
    int K = 2 * C;
    int t = blockIdx.x * blockDim.x + threadIdx.x;
    if (t >= Nv * K) return;
    int v = t / K;
    int k = t - v * K;
    float val = (k < C) ? degw[v] * h[(size_t)v * C + k]
                        : agg[(size_t)v * C + (k - C)];
    xin[t] = (bf16)val;
}

// ---------------------------------------------------------------------------
// VT[f][k] (Fpad x K, row major, bf16) from W (K x F, f32):
//   k <  C : W_top - W_bot   (folds the "concat([v, n-v])" into one GEMM)
//   k >= C : W_bot
// rows f >= F zero-padded.
// ---------------------------------------------------------------------------
__global__ void k_prep_w(const float* __restrict__ W, bf16* __restrict__ VT,
                         int C, int F, int Fpad) {
    int K = 2 * C;
    int t = blockIdx.x * blockDim.x + threadIdx.x;
    if (t >= Fpad * K) return;
    int f = t / K;
    int k = t - f * K;
    float val = 0.0f;
    if (f < F)
        val = (k < C) ? (W[(size_t)k * F + f] - W[(size_t)(k + C) * F + f])
                      : W[(size_t)k * F + f];
    VT[t] = (bf16)val;
}

// ---------------------------------------------------------------------------
// WMMA GEMM + epilogue, register-blocked along N:
//   each wave computes one 16-row M-tile x (NT*16) output columns.
//   A fragment loaded ONCE per k-step and reused across NT column tiles
//   (NT accumulators live in VGPRs: NT*8 f32 regs).
//   out[r][c] = act( Xin[r][:] . VT[c][:] + degw[r]*bias[c] )
// OOB M-rows: clamp the A address (their D rows are never stored), so EXEC
// stays all-1s at the WMMA (ISA requirement).
// ---------------------------------------------------------------------------
template <int NT>
__global__ __launch_bounds__(256)
void k_wmma_gemm(const bf16* __restrict__ xin, const bf16* __restrict__ VT,
                 const float* __restrict__ degw, const float* __restrict__ bias,
                 float* __restrict__ out, int M, int K, int Fvalid, int ldOut,
                 int actFlag) {
    const int lane = threadIdx.x;                // 0..31
    const int wave = threadIdx.y;                // 0..7
    const int mt   = blockIdx.x * 8 + wave;      // one M-tile per wave
    const int numMt = (M + 15) >> 4;
    if (mt >= numMt) return;                     // wave-uniform exit

    const int m     = lane & 15;
    const int khalf = lane >> 4;

    int rowA = mt * 16 + m;
    if (rowA >= M) rowA = M - 1;                 // clamp: those D rows unused

    // A per-lane: two 8-elem chunks at k = kk + khalf*8 and kk + 16 + khalf*8
    const bf16* aBase = xin + (size_t)rowA * K + khalf * 8;
    // B per-lane: two 8-elem chunks at k = kk + khalf*8 and kk + 16 + khalf*8 for column (nt*16+m)
    const bf16* bBase = VT + (size_t)m * K + khalf * 8;

    union V16 { bf16x16 v; bf16x8 h[2]; };

    v8f acc[NT];
#pragma unroll
    for (int j = 0; j < NT; ++j)
        acc[j] = v8f{0.f, 0.f, 0.f, 0.f, 0.f, 0.f, 0.f, 0.f};

    for (int kk = 0; kk < K; kk += 32) {
        V16 a;
        a.h[0] = *reinterpret_cast<const bf16x8*>(aBase + kk);
        a.h[1] = *reinterpret_cast<const bf16x8*>(aBase + kk + 16);
#pragma unroll
        for (int j = 0; j < NT; ++j) {
            const bf16* bp = bBase + (size_t)j * 16 * K + kk;
            V16 b;
            b.h[0] = *reinterpret_cast<const bf16x8*>(bp);
            b.h[1] = *reinterpret_cast<const bf16x8*>(bp + 16);
            acc[j] = __builtin_amdgcn_wmma_f32_16x16x32_bf16(
                /*neg_a=*/false, a.v, /*neg_b=*/false, b.v,
                /*c_mod=*/(short)0, acc[j], /*reuse_a=*/false, /*reuse_b=*/false);
        }
    }

    // C/D layout: element i -> row (i + 8*khalf), col (lane & 15)
#pragma unroll
    for (int j = 0; j < NT; ++j) {
        const int col = j * 16 + (lane & 15);
        if (col < Fvalid) {
            const float bcol = bias[col];
#pragma unroll
            for (int i = 0; i < 8; ++i) {
                const int r = mt * 16 + khalf * 8 + i;
                if (r < M) {
                    float v = acc[j][i] + degw[r] * bcol;
                    if (actFlag) v = (v > 0.f) ? v : SLOPE * v;
                    out[(size_t)r * ldOut + col] = v;
                }
            }
        }
    }
}

// ---------------------------------------------------------------------------
// Mean pooling: sum + count via atomics, then divide.
// ---------------------------------------------------------------------------
__global__ void k_pool_sum(const float* __restrict__ h, const int* __restrict__ pm,
                           float* __restrict__ sum, float* __restrict__ cnt,
                           int Nv, int C) {
    int c4 = C >> 2;
    int t = blockIdx.x * blockDim.x + threadIdx.x;
    if (t >= Nv * c4) return;
    int v = t / c4;
    int c = (t - v * c4) * 4;
    int u = pm[v];
    const float4 hv = *reinterpret_cast<const float4*>(h + (size_t)v * C + c);
    float* d = sum + (size_t)u * C + c;
    atomicAdd(d + 0, hv.x);
    atomicAdd(d + 1, hv.y);
    atomicAdd(d + 2, hv.z);
    atomicAdd(d + 3, hv.w);
    if (c == 0) atomicAdd(&cnt[u], 1.0f);
}

__global__ void k_pool_div(float* __restrict__ hc, const float* __restrict__ cnt,
                           int Nc, int C) {
    int t = blockIdx.x * blockDim.x + threadIdx.x;
    if (t >= Nc * C) return;
    int u = t / C;
    hc[t] /= fmaxf(cnt[u], 1.0f);
}

// ---------------------------------------------------------------------------
// out[v] = concat( hc[pm[v]][0..C-1], skip[v][0..C-1] )   (N x 2C)
// ---------------------------------------------------------------------------
__global__ void k_unpool_cat(const float* __restrict__ hc, const float* __restrict__ skip,
                             const int* __restrict__ pm, float* __restrict__ out,
                             int Nv, int C) {
    int K = 2 * C;
    int t = blockIdx.x * blockDim.x + threadIdx.x;
    if (t >= Nv * K) return;
    int v = t / K;
    int k = t - v * K;
    out[t] = (k < C) ? hc[(size_t)pm[v] * C + k]
                     : skip[(size_t)v * C + (k - C)];
}

// ---------------------------------------------------------------------------
// Host orchestration
// ---------------------------------------------------------------------------
extern "C" void kernel_launch(void* const* d_in, const int* in_sizes, int n_in,
                              void* d_out, int out_size, void* d_ws, size_t ws_size,
                              hipStream_t stream) {
    (void)in_sizes; (void)n_in; (void)out_size; (void)ws_size;

    const float* x    = (const float*)d_in[0];
    const int*   ea_f = (const int*)  d_in[1];
    const int*   eb_f = (const int*)  d_in[2];
    const float* w_f  = (const float*)d_in[3];
    const int*   ea_c = (const int*)  d_in[4];
    const int*   eb_c = (const int*)  d_in[5];
    const float* w_c  = (const float*)d_in[6];
    const int*   pm   = (const int*)  d_in[7];

    const float* Wl[13];
    const float* Bl[13];
    for (int i = 0; i < 13; ++i) {
        Wl[i] = (const float*)d_in[8 + 2 * i];
        Bl[i] = (const float*)d_in[9 + 2 * i];
    }

    // Layer configs
    int Cc[13], Ff[13], Fp[13];
    for (int i = 0; i < 13; ++i) { Cc[i] = 128; Ff[i] = 128; Fp[i] = 128; }
    Cc[0] = 16;                // first layer: F_IN=16 channels
    Cc[8] = 256;               // skip-concat layer: 256 channels in
    Ff[12] = 3; Fp[12] = 16;   // final layer: 3 outputs, pad to one N-tile

    // ---- carve workspace (256B aligned chunks) ----
    char* p = (char*)d_ws;
    auto alloc = [&](size_t bytes) -> void* {
        void* r = (void*)p;
        p += (bytes + 255) & ~(size_t)255;
        return r;
    };
    float* bufA  = (float*)alloc((size_t)N0 * 256 * sizeof(float));
    float* bufB  = (float*)alloc((size_t)N0 * 256 * sizeof(float));
    float* bufC  = (float*)alloc((size_t)N0 * 128 * sizeof(float));   // skip
    float* cA    = (float*)alloc((size_t)N1 * 128 * sizeof(float));
    float* cB    = (float*)alloc((size_t)N1 * 128 * sizeof(float));
    float* agg   = (float*)alloc((size_t)N0 * 256 * sizeof(float));
    bf16*  xin   = (bf16*) alloc((size_t)N0 * 512 * sizeof(bf16));
    float* degwF = (float*)alloc((size_t)N0 * sizeof(float));
    float* degwC = (float*)alloc((size_t)N1 * sizeof(float));
    float* cnt   = (float*)alloc((size_t)N1 * sizeof(float));
    bf16* VT[13];
    for (int i = 0; i < 13; ++i)
        VT[i] = (bf16*)alloc((size_t)Fp[i] * 2 * Cc[i] * sizeof(bf16));

    // ---- per-edge-set weighted degrees ----
    hipMemsetAsync(degwF, 0, (size_t)N0 * sizeof(float), stream);
    hipMemsetAsync(degwC, 0, (size_t)N1 * sizeof(float), stream);
    k_degw<<<ceil_div(E0, 256), 256, 0, stream>>>(ea_f, w_f, degwF, E0);
    k_degw<<<ceil_div(E1, 256), 256, 0, stream>>>(ea_c, w_c, degwC, E1);

    // ---- prepare folded bf16 weights VT[f][k] ----
    for (int i = 0; i < 13; ++i) {
        int T = Fp[i] * 2 * Cc[i];
        k_prep_w<<<ceil_div(T, 256), 256, 0, stream>>>(Wl[i], VT[i], Cc[i], Ff[i], Fp[i]);
    }

    // ---- one edge-conv layer ----
    auto conv = [&](const float* hin, int Nv, int C,
                    const int* ea, const int* eb, const float* w, int E,
                    const float* degw, const bf16* vt, const float* bias,
                    int Fpad, int Fvalid, float* out, int ldOut, bool act) {
        const int K = 2 * C;
        hipMemsetAsync(agg, 0, (size_t)Nv * C * sizeof(float), stream);
        {
            int T = E * (C / 4);
            k_edge_agg<<<ceil_div(T, 256), 256, 0, stream>>>(ea, eb, w, hin, agg, E, C);
        }
        {
            int T = Nv * K;
            k_build_xin<<<ceil_div(T, 256), 256, 0, stream>>>(hin, agg, degw, xin, Nv, C);
        }
        {
            int numMt = ceil_div(Nv, 16);
            dim3 blk(32, 8);
            int grd = ceil_div(numMt, 8);
            if (Fpad == 128)
                k_wmma_gemm<8><<<grd, blk, 0, stream>>>(
                    xin, vt, degw, bias, out, Nv, K, Fvalid, ldOut, act ? 1 : 0);
            else
                k_wmma_gemm<1><<<grd, blk, 0, stream>>>(
                    xin, vt, degw, bias, out, Nv, K, Fvalid, ldOut, act ? 1 : 0);
        }
    };

    // ---- encoder, fine level ----
    conv(x,    N0, 16,  ea_f, eb_f, w_f, E0, degwF, VT[0], Bl[0], 128, 128, bufA, 128, true);
    conv(bufA, N0, 128, ea_f, eb_f, w_f, E0, degwF, VT[1], Bl[1], 128, 128, bufB, 128, true);
    conv(bufB, N0, 128, ea_f, eb_f, w_f, E0, degwF, VT[2], Bl[2], 128, 128, bufA, 128, true);
    conv(bufA, N0, 128, ea_f, eb_f, w_f, E0, degwF, VT[3], Bl[3], 128, 128, bufC, 128, true); // skip

    // ---- mean pool fine -> coarse ----
    hipMemsetAsync(cA,  0, (size_t)N1 * 128 * sizeof(float), stream);
    hipMemsetAsync(cnt, 0, (size_t)N1 * sizeof(float), stream);
    k_pool_sum<<<ceil_div(N0 * 32, 256), 256, 0, stream>>>(bufC, pm, cA, cnt, N0, 128);
    k_pool_div<<<ceil_div(N1 * 128, 256), 256, 0, stream>>>(cA, cnt, N1, 128);

    // ---- bottom, coarse level ----
    conv(cA, N1, 128, ea_c, eb_c, w_c, E1, degwC, VT[4], Bl[4], 128, 128, cB, 128, true);
    conv(cB, N1, 128, ea_c, eb_c, w_c, E1, degwC, VT[5], Bl[5], 128, 128, cA, 128, true);
    conv(cA, N1, 128, ea_c, eb_c, w_c, E1, degwC, VT[6], Bl[6], 128, 128, cB, 128, true);
    conv(cB, N1, 128, ea_c, eb_c, w_c, E1, degwC, VT[7], Bl[7], 128, 128, cA, 128, true);

    // ---- unpool + concat skip -> 256-channel fine features ----
    k_unpool_cat<<<ceil_div(N0 * 256, 256), 256, 0, stream>>>(cA, bufC, pm, bufA, N0, 128);

    // ---- decoder, fine level ----
    conv(bufA, N0, 256, ea_f, eb_f, w_f, E0, degwF, VT[8],  Bl[8],  128, 128, bufB, 128, true);
    conv(bufB, N0, 128, ea_f, eb_f, w_f, E0, degwF, VT[9],  Bl[9],  128, 128, bufA, 128, true);
    conv(bufA, N0, 128, ea_f, eb_f, w_f, E0, degwF, VT[10], Bl[10], 128, 128, bufB, 128, true);
    conv(bufB, N0, 128, ea_f, eb_f, w_f, E0, degwF, VT[11], Bl[11], 128, 128, bufA, 128, true);

    // ---- final layer: 3 outputs, no activation, into d_out ----
    conv(bufA, N0, 128, ea_f, eb_f, w_f, E0, degwF, VT[12], Bl[12], 16, 3,
         (float*)d_out, 3, false);
}
